// RNN_944892805436
// MI455X (gfx1250) — matrix operand order repaired and checked by reading, not compiled, not run on hardware
//
#include <hip/hip_runtime.h>
#include <hip/hip_bf16.h>

typedef __bf16 bf16_t;
typedef bf16_t v16bf __attribute__((ext_vector_type(16)));
typedef float  v8f   __attribute__((ext_vector_type(8)));
typedef unsigned int u32x4 __attribute__((ext_vector_type(4)));
typedef int          i32x8 __attribute__((ext_vector_type(8)));
typedef int          i32x4 __attribute__((ext_vector_type(4)));

#define FRAMES 512
#define BATCH  2048
#define IN_DIM 84
#define HIDDEN 300
#define OUT_DIM 10

#define KP_I 96     // in_dim padded to 3 k-tiles of 32
#define HP   320    // hidden padded (K dim) to 10 k-tiles of 32
#define NP   304    // hidden padded (N dim) to 19 n-tiles of 16
#define NTILES 19
#define KT_H 10
#define KT_I 3
#define KT_ALL (KT_I + KT_H)   // 13

// LDS layout (bytes)
#define WHB_BYTES (NP * HP * 2)             // 194560
#define WIB_BYTES (NP * KP_I * 2)           // 58368
#define W_BYTES   (WHB_BYTES + WIB_BYTES)   // 252928
#define WV_OFF    W_BYTES
#define H_BYTES   (16 * HP * 2)             // 10240
#define X_BYTES   (16 * KP_I * 2)           // 3072 (x tile is double buffered)
#define WV_BYTES  (H_BYTES + 2 * X_BYTES)   // 16384
#define SMEM_TOTAL (WV_OFF + 4 * WV_BYTES)  // 318464 <= 320KB

static_assert(SMEM_TOTAL <= 320 * 1024, "LDS budget");
static_assert(W_BYTES % 8 == 0, "TDM 8B units");

// ---------------- fast tanh (branch-free; HW v_tanh_f32 if available) -----
__device__ __forceinline__ float fast_tanh(float x) {
#if __has_builtin(__builtin_amdgcn_tanhf)
    return __builtin_amdgcn_tanhf(x);
#elif __has_builtin(__builtin_amdgcn_tanh_f32)
    return __builtin_amdgcn_tanh_f32(x);
#else
    // tanh(x) = 1 - 2/(exp2(2*log2e*x)+1); saturates correctly at +-inf
    float e = __builtin_amdgcn_exp2f(2.885390081777927f * x);
    return 1.0f - 2.0f * __builtin_amdgcn_rcpf(e + 1.0f);
#endif
}

// ---------------- fragment loaders (ISA 7.12.2 layouts) ----------------

// 16-bit A matrix 16x32 (MxK): lane l holds row m=l&15;
// halves 0..7 = K[k0+8*hi .. +7], halves 8..15 = K[k0+16+8*hi .. +7]
__device__ __forceinline__ v16bf load_a_frag(const bf16_t* X, int stride, int k0, int lane) {
    const int m  = lane & 15;
    const int hi = lane >> 4;
    const char* p = (const char*)(X + m * stride + k0 + 8 * hi);
    union { uint4 q[2]; v16bf v; } u;
    u.q[0] = *(const uint4*)p;
    u.q[1] = *(const uint4*)(p + 32);
    return u.v;
}

// 16-bit B matrix 32x16 (KxN) stored N-major as W[n][k]:
// lane l holds col n=l&15, halves 0..7 = K[k0+8*hi .. +7], halves 8..15 = K[k0+16+8*hi .. +7] in W row n.
__device__ __forceinline__ v16bf load_b_frag(const bf16_t* W, int stride, int ntile, int k0, int lane) {
    const int n = ntile * 16 + (lane & 15);
    const int k = k0 + ((lane >> 4) << 3);
    const char* p = (const char*)(W + n * stride + k);
    union { uint4 q[2]; v16bf v; } u;
    u.q[0] = *(const uint4*)p;
    u.q[1] = *(const uint4*)(p + 32);
    return u.v;
}

// Unified B-fragment index kt in [0,13): 0..2 -> Wi, 3..12 -> Wh
__device__ __forceinline__ v16bf load_b13(const bf16_t* wib, const bf16_t* whb,
                                          int nt, int kt, int lane) {
    return (kt < KT_I) ? load_b_frag(wib, KP_I, nt, kt * 32, lane)
                       : load_b_frag(whb, HP, nt, (kt - KT_I) * 32, lane);
}

// ---------------- kernel 0: pack weights to padded bf16 ----------------
__global__ void rnn_prep_kernel(const float* __restrict__ Wi, const float* __restrict__ Wh,
                                bf16_t* __restrict__ whb, bf16_t* __restrict__ wib) {
    int t = blockIdx.x * blockDim.x + threadIdx.x;
    if (t < NP * HP) {
        int n = t / HP, k = t % HP;
        whb[t] = (n < HIDDEN && k < HIDDEN) ? (bf16_t)Wh[n * HIDDEN + k] : (bf16_t)0.0f;
    } else {
        t -= NP * HP;
        if (t < NP * KP_I) {
            int n = t / KP_I, k = t % KP_I;
            wib[t] = (n < HIDDEN && k < IN_DIM) ? (bf16_t)Wi[n * IN_DIM + k] : (bf16_t)0.0f;
        }
    }
}

// ---------------- kernel 1: persistent fused RNN ----------------
// 32 blocks x 128 threads (4 waves); each wave owns 16 batch rows end-to-end.
__global__ void __launch_bounds__(128, 1)
rnn_main_kernel(const float* __restrict__ hand,
                const float* __restrict__ h0,
                const bf16_t* __restrict__ wbase,  // whb | wib contiguous in ws
                float* __restrict__ hout) {
    extern __shared__ char smem[];
    const int tid  = threadIdx.x;
    const int lane = tid & 31;
    const int wave = tid >> 5;

    // ---- Stage both weight matrices into LDS: one TDM bulk transfer ----
#if __has_builtin(__builtin_amdgcn_tensor_load_to_lds)
    if (wave == 0) {
        unsigned long long ga = (unsigned long long)(uintptr_t)wbase;
        unsigned lds_addr = (unsigned)(uintptr_t)smem;
        const unsigned NQ = W_BYTES / 8;   // 31616 8-byte elements (1-D tile)
        u32x4 g0;
        g0[0] = 1u;                                               // count=1
        g0[1] = lds_addr;                                         // lds_addr
        g0[2] = (unsigned)ga;                                     // global_addr lo
        g0[3] = (unsigned)((ga >> 32) & 0x01FFFFFFu) | (2u << 30);// addr hi | type=2
        i32x8 g1;
        g1[0] = (int)(3u << 16);                   // data_size = 8B
        g1[1] = (int)((NQ & 0xFFFFu) << 16);       // tensor_dim0[15:0]
        g1[2] = (int)((NQ >> 16) | (1u << 16));    // tensor_dim0[31:16] | tensor_dim1=1
        g1[3] = (int)((NQ & 0xFFFFu) << 16);       // tile_dim0 = NQ
        g1[4] = 1;                                 // tile_dim1 = 1
        g1[5] = (int)NQ;                           // tensor_dim0_stride
        g1[6] = 0;
        g1[7] = 0;
        i32x4 z4 = {0, 0, 0, 0};
        i32x8 z8 = {0, 0, 0, 0, 0, 0, 0, 0};
        __builtin_amdgcn_tensor_load_to_lds(g0, g1, z4, z4, z8, 0);
#if __has_builtin(__builtin_amdgcn_s_wait_tensorcnt)
        __builtin_amdgcn_s_wait_tensorcnt(0);
#else
        asm volatile("s_wait_tensorcnt 0x0" ::: "memory");
#endif
    }
#else
    {
        const uint4* src = (const uint4*)wbase;
        uint4* dst = (uint4*)smem;
        for (int i = tid; i < W_BYTES / 16; i += 128) dst[i] = src[i];
    }
#endif
    __syncthreads();

    bf16_t* whb  = (bf16_t*)smem;
    bf16_t* wib  = (bf16_t*)(smem + WHB_BYTES);
    bf16_t* hbuf = (bf16_t*)(smem + WV_OFF + wave * WV_BYTES);
    bf16_t* xb0  = (bf16_t*)(smem + WV_OFF + wave * WV_BYTES + H_BYTES);
    bf16_t* xb1  = (bf16_t*)(smem + WV_OFF + wave * WV_BYTES + H_BYTES + X_BYTES);

    const int rowbase = blockIdx.x * 64 + wave * 16;

    // Init hidden tile from h0 (pad cols >= HIDDEN with zeros).
    for (int i = lane; i < 16 * HP; i += 32) {
        int r = i / HP, c = i - r * HP;
        float v = (c < HIDDEN) ? h0[(rowbase + r) * HIDDEN + c] : 0.0f;
        hbuf[i] = (bf16_t)v;
    }
    // Zero both x tiles once (cols 84..95 stay zero forever).
    for (int i = lane; i < 2 * 16 * KP_I; i += 32) xb0[i] = (bf16_t)0.0f;

    // Stage frame 0 into xb0.
    {
        const float2* s2 = (const float2*)(hand + (long)rowbase * IN_DIM);
        #pragma unroll
        for (int i = 0; i < 21; ++i) {
            int idx = lane + i * 32;            // index into [16][42] float2
            int r = idx / 42, c = idx - r * 42;
            float2 v = s2[idx];
            union { bf16_t h[2]; unsigned u; } p;
            p.h[0] = (bf16_t)v.x; p.h[1] = (bf16_t)v.y;
            *(unsigned*)((char*)xb0 + r * (KP_I * 2) + c * 4) = p.u;
        }
    }

    const int hi  = lane >> 4;
    const int col = lane & 15;

    // Prime the B-fragment pipeline with (nt=0, kt=0).
    v16bf bcur = load_b13(wib, whb, 0, 0, lane);

    for (int f = 0; f < FRAMES; ++f) {
        bf16_t* xcur = (f & 1) ? xb1 : xb0;
        bf16_t* xnxt = (f & 1) ? xb0 : xb1;

        // Load all 13 A fragments into registers BEFORE overwriting hbuf.
        v16bf afr[KT_ALL];
        #pragma unroll
        for (int kt = 0; kt < KT_I; ++kt) afr[kt] = load_a_frag(xcur, KP_I, kt * 32, lane);
        #pragma unroll
        for (int kt = 0; kt < KT_H; ++kt) afr[KT_I + kt] = load_a_frag(hbuf, HP, kt * 32, lane);

        // Issue next frame's global loads now; commit to LDS after the n-loop
        // so the entire WMMA loop hides the HBM latency.
        float2 xreg[21];
        const bool have_next = (f + 1 < FRAMES);
        if (have_next) {
            const float2* s2 = (const float2*)(hand + ((long)(f + 1) * BATCH + rowbase) * IN_DIM);
            #pragma unroll
            for (int i = 0; i < 21; ++i) xreg[i] = s2[lane + i * 32];
        }

        for (int nt = 0; nt < NTILES; ++nt) {
            v8f acc = {};
            // Software-pipelined B fragments: load kt+1 while computing kt.
            #pragma unroll
            for (int kt = 0; kt < KT_ALL - 1; ++kt) {
                v16bf bnext = load_b13(wib, whb, nt, kt + 1, lane);
                acc = __builtin_amdgcn_wmma_f32_16x16x32_bf16(
                        false, afr[kt], false, bcur, (short)0, acc, false, false);
                bcur = bnext;
            }
            // Prefetch first fragment of the next n-tile (wraps to nt=0 for
            // the next frame; weights are frame-invariant) under the tail.
            {
                int nnt = (nt + 1 < NTILES) ? nt + 1 : 0;
                v16bf bfirst = load_b13(wib, whb, nnt, 0, lane);
                acc = __builtin_amdgcn_wmma_f32_16x16x32_bf16(
                        false, afr[KT_ALL - 1], false, bcur, (short)0, acc, false, false);
                bcur = bfirst;
            }

            // h' = tanh(acc): branch-free; padded cols give tanh(0)=0.
            float hv[8];
            #pragma unroll
            for (int r = 0; r < 8; ++r) hv[r] = fast_tanh(acc[r]);
            const int cc = nt * 16 + col;
            #pragma unroll
            for (int r = 0; r < 8; ++r)
                hbuf[(r + 8 * hi) * HP + cc] = (bf16_t)hv[r];
            if (f == FRAMES - 1) {   // uniform scalar branch
                #pragma unroll
                for (int r = 0; r < 8; ++r)
                    if (cc < HIDDEN)
                        hout[(long)(rowbase + r + 8 * hi) * HIDDEN + cc] = hv[r];
            }
        }

        // Commit next frame's x tile (bf16) into the other LDS buffer.
        if (have_next) {
            #pragma unroll
            for (int i = 0; i < 21; ++i) {
                int idx = lane + i * 32;
                int r = idx / 42, c = idx - r * 42;
                union { bf16_t h[2]; unsigned u; } p;
                p.h[0] = (bf16_t)xreg[i].x; p.h[1] = (bf16_t)xreg[i].y;
                *(unsigned*)((char*)xnxt + r * (KP_I * 2) + c * 4) = p.u;
            }
        }
    }
}

// ---------------- kernel 2: out = h_final @ Wo^T ----------------
__global__ void rnn_out_kernel(const float* __restrict__ hfin,
                               const float* __restrict__ Wo,
                               float* __restrict__ out) {
    int t = blockIdx.x * blockDim.x + threadIdx.x;
    if (t >= BATCH * OUT_DIM) return;
    int b = t / OUT_DIM, o = t - b * OUT_DIM;
    float s = 0.0f;
    #pragma unroll 4
    for (int k = 0; k < HIDDEN; ++k) s += hfin[b * HIDDEN + k] * Wo[o * HIDDEN + k];
    out[t] = s;
}

extern "C" void kernel_launch(void* const* d_in, const int* in_sizes, int n_in,
                              void* d_out, int out_size, void* d_ws, size_t ws_size,
                              hipStream_t stream) {
    const float* hand = (const float*)d_in[0];
    const float* h0   = (const float*)d_in[1];
    const float* Wi   = (const float*)d_in[2];
    const float* Wh   = (const float*)d_in[3];
    const float* Wo   = (const float*)d_in[4];

    float* out  = (float*)d_out;                 // [BATCH, OUT_DIM]
    float* hfin = out + BATCH * OUT_DIM;         // [BATCH, HIDDEN]

    bf16_t* wbase = (bf16_t*)d_ws;               // whb [NP][HP] | wib [NP][KP_I]
    bf16_t* whb = wbase;
    bf16_t* wib = wbase + NP * HP;

    int prep_n = NP * HP + NP * KP_I;
    rnn_prep_kernel<<<(prep_n + 255) / 256, 256, 0, stream>>>(Wi, Wh, whb, wib);

    (void)hipFuncSetAttribute((const void*)rnn_main_kernel,
                              hipFuncAttributeMaxDynamicSharedMemorySize, SMEM_TOTAL);
    rnn_main_kernel<<<dim3(32), dim3(128), SMEM_TOTAL, stream>>>(hand, h0, wbase, hfin);

    rnn_out_kernel<<<(BATCH * OUT_DIM + 255) / 256, 256, 0, stream>>>(hfin, Wo, out);
}
